// DGEBlock_89781996356032
// MI455X (gfx1250) — matrix operand order repaired and checked by reading, not compiled, not run on hardware
//
#include <hip/hip_runtime.h>
#include <hip/hip_bf16.h>
#include <math.h>

#define B_   4
#define C_   64
#define N_   512
#define D_   256
#define DK_  64
#define DFF_ 1024

typedef __attribute__((ext_vector_type(16))) __bf16 v16bf;
typedef __attribute__((ext_vector_type(8)))  float  v8f;
typedef __attribute__((ext_vector_type(2)))  float  v2f;

union BFrag { uint4 q[2]; v16bf v; };

__device__ __forceinline__ unsigned short f2bf(float f) {
  unsigned int u = __float_as_uint(f);
  u += 0x7FFFu + ((u >> 16) & 1u);   // round-to-nearest-even
  return (unsigned short)(u >> 16);
}
__device__ __forceinline__ float gelu_exact(float v) {
  return 0.5f * v * (1.0f + erff(v * 0.70710678118654752f));
}

// ---------------- weight pre-transpose to bf16 [col][K] layouts ----------------
__global__ __launch_bounds__(256) void cvt_w1_kernel(const float* __restrict__ W1,
                                                     unsigned short* __restrict__ W1t) {
  int idx = blockIdx.x * 256 + threadIdx.x;        // over D*DFF = 262144
  int k = idx >> 10;                               // 0..255   (D)
  int c = idx & (DFF_ - 1);                        // 0..1023  (DFF)
  W1t[(size_t)c * D_ + k] = f2bf(W1[idx]);
}
__global__ __launch_bounds__(256) void cvt_w2_kernel(const float* __restrict__ W2,
                                                     unsigned short* __restrict__ W2t) {
  int idx = blockIdx.x * 256 + threadIdx.x;        // over DFF*D = 262144
  int k = idx >> 8;                                // 0..1023  (DFF)
  int c = idx & (D_ - 1);                          // 0..255   (D)
  W2t[(size_t)c * DFF_ + k] = f2bf(W2[idx]);
}

// ---------------- Hc = mean over N ----------------
__global__ __launch_bounds__(256) void mean_kernel(const float* __restrict__ x,
                                                   float* __restrict__ Hc) {
  int idx = blockIdx.x * 256 + threadIdx.x;        // over B*C*D = 65536
  int d  = idx & (D_ - 1);
  int bc = idx >> 8;
  const float* p = x + (size_t)bc * N_ * D_ + d;
  float s = 0.f;
  for (int n = 0; n < N_; ++n) s += p[(size_t)n * D_];
  Hc[idx] = s * (1.0f / N_);
}

// ---------------- Q,K, dynamic adjacency, normalize ----------------
__global__ __launch_bounds__(256) void adj_kernel(const float* __restrict__ Hc,
    const float* __restrict__ Astat, const float* __restrict__ Wq,
    const float* __restrict__ Wk, float* __restrict__ Aw, float* __restrict__ Aout) {
  __shared__ float Qs[C_ * DK_];
  __shared__ float Ks[C_ * DK_];
  __shared__ float Ra[C_ * C_];
  __shared__ float dis[C_];
  int b = blockIdx.x, t = threadIdx.x;
  const float* hb = Hc + (size_t)b * C_ * D_;
  for (int ii = 0; ii < 16; ++ii) {
    int idx = t + 256 * ii;                        // 0..4095
    int c = idx >> 6, k = idx & 63;
    const float* hr = hb + c * D_;
    float sq = 0.f, sk = 0.f;
    for (int d = 0; d < D_; ++d) {
      float h = hr[d];
      sq += h * Wq[d * DK_ + k];
      sk += h * Wk[d * DK_ + k];
    }
    Qs[idx] = sq; Ks[idx] = sk;
  }
  __syncthreads();
  for (int ii = 0; ii < 16; ++ii) {
    int idx = t + 256 * ii; int c = idx >> 6, e = idx & 63;
    float s = 0.f;
    for (int k = 0; k < DK_; ++k) s += Qs[c * DK_ + k] * Ks[e * DK_ + k];
    Ra[idx] = Astat[idx] + tanhf(s * 0.125f);      // /(sqrt(64)*tau)
  }
  __syncthreads();
  for (int ii = 0; ii < 16; ++ii) {                // symmetrize + zero diag (reuse Qs)
    int idx = t + 256 * ii; int c = idx >> 6, e = idx & 63;
    Qs[idx] = (c == e) ? 0.f : 0.5f * (Ra[c * 64 + e] + Ra[e * 64 + c]);
  }
  __syncthreads();
  if (t < C_) {
    float d = 0.f;
    for (int e = 0; e < C_; ++e) d += Qs[t * 64 + e];
    dis[t] = rsqrtf(fmaxf(d, 1e-6f));
  }
  __syncthreads();
  for (int ii = 0; ii < 16; ++ii) {
    int idx = t + 256 * ii; int c = idx >> 6, e = idx & 63;
    float av = dis[c] * dis[e] * Qs[idx];
    Aw[b * 4096 + idx]   = av;
    Aout[b * 4096 + idx] = av;                     // third output
  }
}

// ---------------- Hprop = A@x (fp32 WMMA 16x16x4), +x, LN1 ----------------
__global__ __launch_bounds__(256) void hprop_ln1_kernel(const float* __restrict__ x,
    const float* __restrict__ Aw, const float* __restrict__ g1,
    const float* __restrict__ be1, float* __restrict__ H1,
    unsigned short* __restrict__ H1bf) {
  __shared__ float S[C_ * D_];                     // 64 KB: full (c,d) tile for this (b,n)
  int bid = blockIdx.x;
  int b = bid >> 9, n = bid & (N_ - 1);
  int w = threadIdx.x >> 5, lane = threadIdx.x & 31;
  int r = lane & 15, h = lane >> 4;
  const float* Ab = Aw + b * C_ * C_;
  int c0 = (w >> 1) * 16;                          // m-tile per wave pair
  float a0[16], a1[16];                            // A fragments, full K=64
#pragma unroll
  for (int k = 0; k < 16; ++k) {
    a0[k] = Ab[(c0 + r) * C_ + 4 * k + 2 * h];
    a1[k] = Ab[(c0 + r) * C_ + 4 * k + 2 * h + 1];
  }
  const size_t ND = (size_t)N_ * D_;
  const float* xb = x + (size_t)b * C_ * ND + (size_t)n * D_;
#pragma unroll
  for (int i = 0; i < 8; ++i) {
    int d0 = ((w & 1) * 8 + i) * 16;
    // preload ALL B fragments + residual so loads issue as one long clause
    float b0[16], b1[16], resv[8];
#pragma unroll
    for (int k = 0; k < 16; ++k) {
      const float* xe = xb + (size_t)(4 * k + 2 * h) * ND + d0 + r;
      b0[k] = xe[0];
      b1[k] = xe[ND];
    }
#pragma unroll
    for (int j = 0; j < 8; ++j)
      resv[j] = xb[(size_t)(c0 + j + h * 8) * ND + d0 + r];
    v8f acc = {};
#pragma unroll
    for (int k = 0; k < 16; ++k) {
      v2f av; av[0] = a0[k]; av[1] = a1[k];
      v2f bv; bv[0] = b0[k]; bv[1] = b1[k];
      acc = __builtin_amdgcn_wmma_f32_16x16x4_f32(false, av, false, bv,
                                                  (short)0, acc, false, false);
    }
#pragma unroll
    for (int j = 0; j < 8; ++j) {
      int row = c0 + j + h * 8;
      S[row * D_ + d0 + r] = acc[j] + resv[j];     // residual
    }
  }
  __syncthreads();
  // LayerNorm each channel row (wave per row, 8 rows per wave)
  for (int q = 0; q < 8; ++q) {
    int c = w * 8 + q;
    float s = 0.f, ss = 0.f;
#pragma unroll
    for (int t2 = 0; t2 < 8; ++t2) {
      float v = S[c * D_ + lane + 32 * t2];
      s += v; ss += v * v;
    }
#pragma unroll
    for (int m = 16; m >= 1; m >>= 1) {
      s  += __shfl_xor(s,  m, 32);
      ss += __shfl_xor(ss, m, 32);
    }
    float mean = s * (1.f / D_);
    float var  = ss * (1.f / D_) - mean * mean;
    float rs   = rsqrtf(var + 1e-5f);
    size_t rowg = ((size_t)(b * C_ + c) * N_ + n) * D_;
#pragma unroll
    for (int t2 = 0; t2 < 8; ++t2) {
      int col = lane + 32 * t2;
      float hv = (S[c * D_ + col] - mean) * rs * g1[col] + be1[col];
      H1[rowg + col]   = hv;
      H1bf[rowg + col] = f2bf(hv);
    }
  }
}

// ---------------- FFN GEMM1: G = gelu(H1 @ W1 + b1), bf16 WMMA ----------------
__global__ __launch_bounds__(256) void ffn1_kernel(const unsigned short* __restrict__ H1bf,
    const unsigned short* __restrict__ W1t, const float* __restrict__ b1,
    unsigned short* __restrict__ Gbf) {
  int w = threadIdx.x >> 5, lane = threadIdx.x & 31;
  int r = lane & 15, h = lane >> 4;
  int gw = blockIdx.x * 8 + w;                     // 131072 wave jobs
  int mt = gw >> 4, ng = gw & 15;
  int row0 = mt * 16, c0 = ng * 64;
  v16bf afr[8];                                    // full K=256 A fragments
  const unsigned short* arow = H1bf + (size_t)(row0 + r) * D_;
#pragma unroll
  for (int kf = 0; kf < 8; ++kf) {
    BFrag u;
    u.q[0] = *reinterpret_cast<const uint4*>(arow + kf * 32 + h * 8);
    u.q[1] = *reinterpret_cast<const uint4*>(arow + kf * 32 + 16 + h * 8);
    afr[kf] = u.v;
  }
#pragma unroll
  for (int nt = 0; nt < 4; ++nt) {
    int colb = c0 + nt * 16;
    const unsigned short* bcol = W1t + (size_t)(colb + r) * D_;
    // stage all 8 B fragments first, then run the WMMA chain back-to-back
    BFrag ub[8];
#pragma unroll
    for (int kf = 0; kf < 8; ++kf) {
      ub[kf].q[0] = *reinterpret_cast<const uint4*>(bcol + kf * 32 + h * 8);
      ub[kf].q[1] = *reinterpret_cast<const uint4*>(bcol + kf * 32 + 16 + h * 8);
    }
    v8f acc = {};
#pragma unroll
    for (int kf = 0; kf < 8; ++kf)
      acc = __builtin_amdgcn_wmma_f32_16x16x32_bf16(false, afr[kf], false, ub[kf].v,
                                                    (short)0, acc, false, false);
    int colx = colb + r;
    float bias = b1[colx];
#pragma unroll
    for (int j = 0; j < 8; ++j) {
      int row = row0 + j + h * 8;
      Gbf[(size_t)row * DFF_ + colx] = f2bf(gelu_exact(acc[j] + bias));
    }
  }
}

// ---------------- FFN GEMM2: S2 = G @ W2 + b2 + H1 ----------------
__global__ __launch_bounds__(256) void ffn2_kernel(const unsigned short* __restrict__ Gbf,
    const unsigned short* __restrict__ W2t, const float* __restrict__ b2,
    const float* __restrict__ H1, float* __restrict__ S2) {
  int w = threadIdx.x >> 5, lane = threadIdx.x & 31;
  int r = lane & 15, h = lane >> 4;
  int gw = blockIdx.x * 8 + w;                     // 32768 wave jobs
  int mt = gw >> 2, ng = gw & 3;
  int row0 = mt * 16, c0 = ng * 64;
  v8f acc[4] = {{}, {}, {}, {}};
  const unsigned short* arow = Gbf + (size_t)(row0 + r) * DFF_;
#pragma unroll 4
  for (int kf = 0; kf < 32; ++kf) {
    // stream-prefetch the A row four chunks ahead (G is 268MB > L2)
    if (kf + 4 < 32)
      __builtin_prefetch(arow + (kf + 4) * 32 + h * 8, 0, 1);
    BFrag ua;
    ua.q[0] = *reinterpret_cast<const uint4*>(arow + kf * 32 + h * 8);
    ua.q[1] = *reinterpret_cast<const uint4*>(arow + kf * 32 + 16 + h * 8);
    BFrag ub[4];
#pragma unroll
    for (int nt = 0; nt < 4; ++nt) {
      const unsigned short* bcol =
          W2t + (size_t)(c0 + nt * 16 + r) * DFF_ + kf * 32 + h * 8;
      ub[nt].q[0] = *reinterpret_cast<const uint4*>(bcol);
      ub[nt].q[1] = *reinterpret_cast<const uint4*>(bcol + 16);
    }
#pragma unroll
    for (int nt = 0; nt < 4; ++nt)
      acc[nt] = __builtin_amdgcn_wmma_f32_16x16x32_bf16(false, ua.v, false, ub[nt].v,
                                                        (short)0, acc[nt], false, false);
  }
#pragma unroll
  for (int nt = 0; nt < 4; ++nt) {
    int colx = c0 + nt * 16 + r;
    float bias = b2[colx];
#pragma unroll
    for (int j = 0; j < 8; ++j) {
      size_t row = (size_t)row0 + j + h * 8;
      S2[row * D_ + colx] = acc[nt][j] + bias + H1[row * D_ + colx];
    }
  }
}

// ---------------- LN2 -> H_out ----------------
__global__ __launch_bounds__(256) void ln2_kernel(const float* __restrict__ S2,
    const float* __restrict__ g2, const float* __restrict__ be2,
    float* __restrict__ out) {
  int w = threadIdx.x >> 5, lane = threadIdx.x & 31;
  size_t row = (size_t)blockIdx.x * 8 + w;
  const float* p = S2 + row * D_;
  float s = 0.f, ss = 0.f, v[8];
#pragma unroll
  for (int t = 0; t < 8; ++t) {
    v[t] = p[lane + 32 * t];
    s += v[t]; ss += v[t] * v[t];
  }
#pragma unroll
  for (int m = 16; m >= 1; m >>= 1) {
    s  += __shfl_xor(s,  m, 32);
    ss += __shfl_xor(ss, m, 32);
  }
  float mean = s * (1.f / D_);
  float var  = ss * (1.f / D_) - mean * mean;
  float rs   = rsqrtf(var + 1e-5f);
  float* o = out + row * D_;
#pragma unroll
  for (int t = 0; t < 8; ++t) {
    int col = lane + 32 * t;
    o[col] = (v[t] - mean) * rs * g2[col] + be2[col];
  }
}

extern "C" void kernel_launch(void* const* d_in, const int* in_sizes, int n_in,
                              void* d_out, int out_size, void* d_ws, size_t ws_size,
                              hipStream_t stream) {
  const float* x     = (const float*)d_in[0];
  const float* Astat = (const float*)d_in[1];
  const float* Wq    = (const float*)d_in[2];
  const float* Wk    = (const float*)d_in[3];
  const float* W1    = (const float*)d_in[4];
  const float* b1    = (const float*)d_in[5];
  const float* W2    = (const float*)d_in[6];
  const float* b2    = (const float*)d_in[7];
  const float* g1    = (const float*)d_in[8];
  const float* be1   = (const float*)d_in[9];
  const float* g2    = (const float*)d_in[10];
  const float* be2   = (const float*)d_in[11];

  char* ws = (char*)d_ws;
  float*          Hc   = (float*)(ws + 0);                      // 262144 B
  float*          Aw   = (float*)(ws + 262144);                 // 65536 B
  float*          H1   = (float*)(ws + 327680);                 // 134217728 B
  unsigned short* H1bf = (unsigned short*)(ws + 134545408ULL);  // 67108864 B
  unsigned short* Gbf  = (unsigned short*)(ws + 201654272ULL);  // 268435456 B
  unsigned short* W1t  = (unsigned short*)(ws + 470089728ULL);  // 524288 B
  unsigned short* W2t  = (unsigned short*)(ws + 470614016ULL);  // 524288 B

  float* out  = (float*)d_out;                  // H_out: 33554432 floats
  float* S2   = out + 33554432;                 // x-section reused as pre-LN scratch
  float* Aout = out + 67108864;                 // A: 16384 floats

  cvt_w1_kernel<<<1024, 256, 0, stream>>>(W1, W1t);
  cvt_w2_kernel<<<1024, 256, 0, stream>>>(W2, W2t);
  mean_kernel<<<256, 256, 0, stream>>>(x, Hc);
  adj_kernel<<<B_, 256, 0, stream>>>(Hc, Astat, Wq, Wk, Aw, Aout);
  hprop_ln1_kernel<<<B_ * N_, 256, 0, stream>>>(x, Aw, g1, be1, H1, H1bf);
  ffn1_kernel<<<16384, 256, 0, stream>>>(H1bf, W1t, b1, Gbf);
  ffn2_kernel<<<4096, 256, 0, stream>>>(Gbf, W2t, b2, H1, S2);
  ln2_kernel<<<16384, 256, 0, stream>>>(S2, g2, be2, out);
  // second output: copy x into its section (after ln2 has consumed the scratch)
  hipMemcpyAsync(out + 33554432, x, (size_t)33554432 * 4,
                 hipMemcpyDeviceToDevice, stream);
}
